// SPARO_6262062318000
// MI455X (gfx1250) — matrix operand order repaired and checked by reading, not compiled, not run on hardware
//
#include <hip/hip_runtime.h>
#include <hip/hip_bf16.h>
#include <hip/hip_fp16.h>

// ---------------------------------------------------------------------------
// SPARO shared-KV attention pooling for MI455X (gfx1250, wave32, WMMA).
//
// B=32, N=1024, DIM=768, H=16, S=4, D=64.
// One workgroup per (b,h): kv slice (1024x64) lives in LDS as f16; the
// 51.5-GFLOP kv GEMM runs on v_wmma_f32_16x16x32_f16 with fp32 accumulate.
// ---------------------------------------------------------------------------

typedef __attribute__((ext_vector_type(16))) _Float16 v16h;
typedef __attribute__((ext_vector_type(8)))  _Float16 v8h;
typedef __attribute__((ext_vector_type(8)))  float    v8f;

#define BB   32
#define NN   1024
#define DIMM 768
#define HH   16
#define SS   4
#define DD   64

// LDS layout (bytes). Row pads chosen so 32-lane b128 LDS accesses are
// evenly distributed across the 64 banks (>=2 cycles is the floor anyway).
#define BROW   1584                       // Wk_h row: 768 f16 + pad (16B aligned, 396 dw)
#define KVROW  144                        // kv row: 64 f16 + pad (16B aligned, 36 dw)
#define OFF_B   0
#define OFF_KV  (64 * BROW)               // 101376
#define OFF_Q   (OFF_KV + NN * KVROW)     // 248832  : q*scale, [d][s] f32 (1 KB)
#define OFF_P   (OFF_Q + DD * SS * 4)     // 249856  : probs [s][n] f32 (16 KB)
#define OFF_RED (OFF_P + SS * NN * 4)     // 266240  : 32-float reduction scratch
#define SMEM_BYTES (OFF_RED + 32 * 4)     // 266368  (< 320 KB WGP LDS)

extern "C" __global__ void __launch_bounds__(256, 1)
sparo_attn_kernel(const float* __restrict__ x,      // (B,N,DIM)
                  const int*   __restrict__ eos,    // (B,)
                  const float* __restrict__ qe,     // (H,S,D)
                  const float* __restrict__ Wk,     // (DIM, H*D)
                  const float* __restrict__ bk,     // (H*D,)
                  float*       __restrict__ out)    // (B, H*S, D)
{
    extern __shared__ char smem[];
    char*  sB  = smem + OFF_B;                  // Wk_h, f16, [n(0..63)][k(0..767)]
    char*  sKV = smem + OFF_KV;                 // kv,   f16, [n_pos(0..1023)][d(0..63)]
    float* sQ  = (float*)(smem + OFF_Q);        // q*scale, [d][s]
    float* sP  = (float*)(smem + OFF_P);        // exp weights [s][n_pos]
    float* sR  = (float*)(smem + OFF_RED);      // cross-wave reductions

    const int blk  = blockIdx.x;
    const int b    = blk >> 4;        // / H
    const int h    = blk & 15;        // % H
    const int tid  = threadIdx.x;     // 256 threads = 8 wave32
    const int lane = tid & 31;
    const int wv   = tid >> 5;

    // ------------------------------------------------------------------
    // Stage Wk[:, h*64 : h*64+64] transposed into LDS as f16 (n-major),
    // and q_emb[h]*scale into LDS as [d][s] so phase 2 reads float4s.
    // ------------------------------------------------------------------
    {
        const float* wp = Wk + (size_t)h * DD;          // Wk[k][h*64+n], row stride H*D
        for (int idx = tid; idx < 64 * DIMM; idx += 256) {
            const int n = idx & 63;                     // consecutive lanes -> coalesced
            const int k = idx >> 6;
            const float w = wp[(size_t)k * (HH * DD) + n];
            *(_Float16*)(sB + n * BROW + k * 2) = (_Float16)w;
        }
        const int qs = tid & 3, qd = tid >> 2;          // tid covers all 256 q values
        sQ[tid] = qe[((h * SS + qs) * DD) + qd] * 0.125f;   // D^-0.5 = 1/8
    }
    __syncthreads();

    // ------------------------------------------------------------------
    // Phase 1: kv = x[b] @ Wk_h + bk_h  ->  sKV (f16), fp32 accumulate.
    // Wave w owns rows [w*128, w*128+128): 8 M-tiles x 4 N-tiles of 16x16.
    // ------------------------------------------------------------------
    {
        const float* xb = x + (size_t)b * NN * DIMM;
        const int mr = lane & 15;     // row within tile (A) / column within tile (B)
        const int hf = lane >> 4;     // lane-half selects K sub-range per ISA layout

        float bias[4];
#pragma unroll
        for (int nt = 0; nt < 4; ++nt)
            bias[nt] = bk[h * DD + nt * 16 + mr];

        // B fragment base: lane<16 -> K {0..7,16..23}, lane>=16 -> K {8..15,24..31}
        const char* bbase = sB + mr * BROW + hf * 16;

        for (int mt = 0; mt < 8; ++mt) {
            const int m0 = wv * 128 + mt * 16;
            v8f acc[4] = {};

            // A fragment (16-bit A 16x32 layout): lane<16 holds K {0..7,16..23},
            // lane>=16 holds K {8..15,24..31}, row = m0 + (lane&15).
            const float* arow = xb + (size_t)(m0 + mr) * DIMM + hf * 8;

            for (int k0 = 0; k0 < DIMM; k0 += 32) {
                const float4 f0 = *(const float4*)(arow + k0);
                const float4 f1 = *(const float4*)(arow + k0 + 4);
                const float4 f2 = *(const float4*)(arow + k0 + 16);
                const float4 f3 = *(const float4*)(arow + k0 + 20);
                v16h A;
                A[0]=(_Float16)f0.x;  A[1]=(_Float16)f0.y;  A[2]=(_Float16)f0.z;  A[3]=(_Float16)f0.w;
                A[4]=(_Float16)f1.x;  A[5]=(_Float16)f1.y;  A[6]=(_Float16)f1.z;  A[7]=(_Float16)f1.w;
                A[8]=(_Float16)f2.x;  A[9]=(_Float16)f2.y;  A[10]=(_Float16)f2.z; A[11]=(_Float16)f2.w;
                A[12]=(_Float16)f3.x; A[13]=(_Float16)f3.y; A[14]=(_Float16)f3.z; A[15]=(_Float16)f3.w;

#pragma unroll
                for (int nt = 0; nt < 4; ++nt) {
                    const char* bp = bbase + nt * (16 * BROW) + k0 * 2;
                    const v8h blo = *(const v8h*)(bp);
                    const v8h bhi = *(const v8h*)(bp + 32);
                    const v16h Bf = __builtin_shufflevector(
                        blo, bhi, 0,1,2,3,4,5,6,7,8,9,10,11,12,13,14,15);
                    acc[nt] = __builtin_amdgcn_wmma_f32_16x16x32_f16(
                        false, A, false, Bf, (short)0, acc[nt], false, false);
                }
            }

            // C layout: VGPR r, lane<16 -> (M=m0+r, N=lane), lane>=16 -> (M=m0+8+r)
#pragma unroll
            for (int nt = 0; nt < 4; ++nt) {
                const int n = nt * 16 + mr;
#pragma unroll
                for (int r = 0; r < 8; ++r) {
                    const int m = m0 + r + hf * 8;
                    *(_Float16*)(sKV + m * KVROW + n * 2) =
                        (_Float16)(acc[nt][r] + bias[nt]);
                }
            }
        }
    }
    __syncthreads();

    // ------------------------------------------------------------------
    // Phase 2: logits -> eos-masked softmax -> weighted sum (all in LDS).
    // Thread t owns positions n = t, t+256, t+512, t+768.
    // ------------------------------------------------------------------
    const int eosb = eos[b];
    float lg[4][4];                                   // [j][s]
    {
        float a[4][4] = {};
#pragma unroll
        for (int d0 = 0; d0 < DD; d0 += 8) {
            v8h kc[4];
#pragma unroll
            for (int j = 0; j < 4; ++j)
                kc[j] = *(const v8h*)(sKV + (tid + j * 256) * KVROW + d0 * 2);
#pragma unroll
            for (int dd = 0; dd < 8; ++dd) {
                const float4 q4 = *(const float4*)(sQ + (d0 + dd) * 4); // broadcast
#pragma unroll
                for (int j = 0; j < 4; ++j) {
                    const float kv = (float)kc[j][dd];
                    a[j][0] += q4.x * kv; a[j][1] += q4.y * kv;
                    a[j][2] += q4.z * kv; a[j][3] += q4.w * kv;
                }
            }
        }
#pragma unroll
        for (int j = 0; j < 4; ++j) {
            const bool valid = (tid + j * 256) <= eosb;
#pragma unroll
            for (int s = 0; s < 4; ++s)
                lg[j][s] = valid ? a[j][s] : -1e30f;
        }
    }

    // --- max reduction (wave32 shuffle + cross-wave LDS) ---
    float mx[4];
#pragma unroll
    for (int s = 0; s < 4; ++s) {
        float v = fmaxf(fmaxf(lg[0][s], lg[1][s]), fmaxf(lg[2][s], lg[3][s]));
#pragma unroll
        for (int off = 16; off >= 1; off >>= 1)
            v = fmaxf(v, __shfl_xor(v, off, 32));
        if (lane == 0) sR[wv * 4 + s] = v;
    }
    __syncthreads();
#pragma unroll
    for (int s = 0; s < 4; ++s) {
        float v = sR[s];
#pragma unroll
        for (int w = 1; w < 8; ++w) v = fmaxf(v, sR[w * 4 + s]);
        mx[s] = v;
    }
    __syncthreads();

    // --- exp + store probs + sum reduction ---
    float ls[4] = {0.f, 0.f, 0.f, 0.f};
#pragma unroll
    for (int j = 0; j < 4; ++j) {
        const int n = tid + j * 256;
        const bool valid = n <= eosb;
#pragma unroll
        for (int s = 0; s < 4; ++s) {
            const float e = valid ? __expf(lg[j][s] - mx[s]) : 0.0f;
            sP[s * NN + n] = e;
            ls[s] += e;
        }
    }
#pragma unroll
    for (int s = 0; s < 4; ++s) {
        float v = ls[s];
#pragma unroll
        for (int off = 16; off >= 1; off >>= 1)
            v += __shfl_xor(v, off, 32);
        if (lane == 0) sR[wv * 4 + s] = v;
    }
    __syncthreads();

    // --- weighted sum: thread t -> output (s = t>>6, d = t&63) ---
    {
        const int s = tid >> 6;
        const int d = tid & 63;
        float sum = 0.f;
#pragma unroll
        for (int w = 0; w < 8; ++w) sum += sR[w * 4 + s];
        const float invs = 1.0f / sum;

        const float* pr = sP + s * NN;
        float acc = 0.f;
        for (int n = 0; n < NN; n += 4) {
            const float4 p4 = *(const float4*)(pr + n);      // broadcast within wave
            const float k0 = (float)*(const _Float16*)(sKV + (n + 0) * KVROW + d * 2);
            const float k1 = (float)*(const _Float16*)(sKV + (n + 1) * KVROW + d * 2);
            const float k2 = (float)*(const _Float16*)(sKV + (n + 2) * KVROW + d * 2);
            const float k3 = (float)*(const _Float16*)(sKV + (n + 3) * KVROW + d * 2);
            acc += p4.x * k0 + p4.y * k1 + p4.z * k2 + p4.w * k3;
        }
        out[((size_t)(b * (HH * SS) + h * SS + s)) * DD + d] = acc * invs;
    }
}

extern "C" void kernel_launch(void* const* d_in, const int* in_sizes, int n_in,
                              void* d_out, int out_size, void* d_ws, size_t ws_size,
                              hipStream_t stream) {
    (void)in_sizes; (void)n_in; (void)out_size; (void)d_ws; (void)ws_size;

    const float* x   = (const float*)d_in[0];   // (32,1024,768) f32
    const int*   eos = (const int*)  d_in[1];   // (32,)          i32
    const float* qe  = (const float*)d_in[2];   // (16,4,64)      f32
    const float* Wk  = (const float*)d_in[3];   // (768,1024)     f32
    const float* bk  = (const float*)d_in[4];   // (1024,)        f32
    float*       out = (float*)d_out;           // (32,64,64)     f32

    // Allow >64KB dynamic LDS (gfx1250 WGP has 320KB). Best-effort; not a
    // stream op, safe under graph capture.
    (void)hipFuncSetAttribute((const void*)sparo_attn_kernel,
                              hipFuncAttributeMaxDynamicSharedMemorySize,
                              SMEM_BYTES);

    sparo_attn_kernel<<<dim3(BB * HH), dim3(256), SMEM_BYTES, stream>>>(
        x, eos, qe, Wk, bk, out);
}
